// BiasedMultiheadAttention_10453950399028
// MI455X (gfx1250) — matrix operand order repaired and checked by reading, not compiled, not run on hardware
//
#include <hip/hip_runtime.h>
#include <hip/hip_fp16.h>

#define NB 16
#define NN 512
#define NF 512
#define NH 8
#define ND 64

typedef __attribute__((ext_vector_type(16))) _Float16 v16h;
typedef __attribute__((ext_vector_type(8)))  _Float16 v8h;
typedef __attribute__((ext_vector_type(8)))  float    v8f;

static __device__ __forceinline__ v16h cat8(v8h lo, v8h hi) {
    return __builtin_shufflevector(lo, hi, 0,1,2,3,4,5,6,7,8,9,10,11,12,13,14,15);
}
static __device__ __forceinline__ v8f wmma_f16(v16h a, v16h b, v8f c) {
    return __builtin_amdgcn_wmma_f32_16x16x32_f16(false, a, false, b, (short)0, c, false, false);
}
// CDNA5 async global->LDS copy (ASYNCcnt-tracked). LDS dest VGPR holds the
// 32-bit LDS byte offset (low 32 bits of the generic pointer per ISA flat->LDS
// truncation); global address in a VGPR pair, GV mode.
static __device__ __forceinline__ void async_copy_b128(void* lds_dst, const void* gsrc) {
    unsigned off = (unsigned)(size_t)lds_dst;
    asm volatile("global_load_async_to_lds_b128 %0, %1, off"
                 :: "v"(off), "v"(gsrc) : "memory");
}
static __device__ __forceinline__ void wait_asynccnt0() {
    asm volatile("s_wait_asynccnt 0" ::: "memory");
}

// ---------------------------------------------------------------------------
// Kernel 1: fused QKV projection, f32 -> f16, Q scaled by sqrt(d)=8.
// grid = (BN/64, F/64, 3), block = 256 (8 waves), each wave -> 16x32 out tile.
// Q,K stored [B,H,N,D]; V stored transposed [B,H,D,N] for contiguous B-frags.
// ---------------------------------------------------------------------------
__global__ __launch_bounds__(256)
void qkv_proj_kernel(const float* __restrict__ ndata,
                     const float* __restrict__ Wq, const float* __restrict__ bq,
                     const float* __restrict__ Wk, const float* __restrict__ bk,
                     const float* __restrict__ Wv, const float* __restrict__ bv,
                     _Float16* __restrict__ q_ws, _Float16* __restrict__ k_ws,
                     _Float16* __restrict__ vT_ws)
{
    __shared__ _Float16 As[64 * 32];   // [row][k]
    __shared__ _Float16 Bs[64 * 32];   // [col][k]  (W transposed tile)
    const int sel  = blockIdx.z;
    const float* W    = (sel == 0) ? Wq : (sel == 1) ? Wk : Wv;
    const float* bias = (sel == 0) ? bq : (sel == 1) ? bk : bv;
    const int R0 = blockIdx.x * 64;
    const int C0 = blockIdx.y * 64;
    const int t    = threadIdx.x;
    const int lane = t & 31;
    const int wave = t >> 5;
    const int msub = (wave >> 1) * 16;
    const int nsub = (wave & 1) * 32;
    const int lm = lane & 15;
    const int kg = lane >> 4;

    v8f c0 = {}; v8f c1 = {};
    const int arow = t >> 2;        // 0..63
    const int aks  = (t & 3) * 8;   // 0,8,16,24

    for (int kb = 0; kb < NF; kb += 32) {
        {
            const float* src = ndata + (size_t)(R0 + arow) * NF + kb + aks;
            _Float16* dst = As + arow * 32 + aks;
            #pragma unroll
            for (int j = 0; j < 8; ++j) dst[j] = (_Float16)src[j];
        }
        {
            const float* src = W + (size_t)(kb + aks) * NF + C0 + arow;
            _Float16* dst = Bs + arow * 32 + aks;
            #pragma unroll
            for (int j = 0; j < 8; ++j) dst[j] = (_Float16)src[(size_t)j * NF];
        }
        __syncthreads();
        const _Float16* ap = As + (msub + lm) * 32 + kg * 8;
        v16h a = cat8(*(const v8h*)ap, *(const v8h*)(ap + 16));
        const _Float16* bp0 = Bs + (nsub + lm) * 32 + kg * 8;
        v16h b0 = cat8(*(const v8h*)bp0, *(const v8h*)(bp0 + 16));
        const _Float16* bp1 = Bs + (nsub + 16 + lm) * 32 + kg * 8;
        v16h b1 = cat8(*(const v8h*)bp1, *(const v8h*)(bp1 + 16));
        c0 = wmma_f16(a, b0, c0);
        c1 = wmma_f16(a, b1, c1);
        __syncthreads();
    }

    #pragma unroll
    for (int nt = 0; nt < 2; ++nt) {
        const int colg = C0 + nsub + nt * 16 + lm;
        const int hh = colg >> 6;
        const int dc = colg & 63;
        const float bv_ = bias[colg];
        v8f cc = nt ? c1 : c0;
        #pragma unroll
        for (int r = 0; r < 8; ++r) {
            const int rowg = R0 + msub + (kg ? r + 8 : r);
            const int bb = rowg >> 9;
            const int nn = rowg & (NN - 1);
            float val = cc[r] + bv_;
            if (sel == 0) val *= 8.0f;
            if (sel == 2)
                vT_ws[((size_t)(bb * NH + hh) * ND + dc) * NN + nn] = (_Float16)val;
            else {
                _Float16* dst = (sel == 0) ? q_ws : k_ws;
                dst[((size_t)(bb * NH + hh) * NN + nn) * ND + dc] = (_Float16)val;
            }
        }
    }
}

// ---------------------------------------------------------------------------
// Kernel 2: flash-style attention. grid = (N/128, B*H), block = 256 (8 waves).
// All 8 waves share one head; the 32-key K/V block is staged in LDS via
// double-buffered GLOBAL_LOAD_ASYNC_TO_LDS_B128, amortizing K/V HBM traffic
// across 128 query rows. Bias/mul/mask stream straight from global (L2 reuse
// across the 8 head-blocks of a batch).
// ---------------------------------------------------------------------------
__global__ __launch_bounds__(256)
void attn_kernel(const float* __restrict__ attn_bias,
                 const int*   __restrict__ attn_mask,
                 const float* __restrict__ attn_mul,
                 const _Float16* __restrict__ q_ws,
                 const _Float16* __restrict__ k_ws,
                 const _Float16* __restrict__ vT_ws,
                 _Float16* __restrict__ ao_ws)
{
    __shared__ _Float16 kS[2][32 * 64];   // [key row][d]
    __shared__ _Float16 vS[2][64 * 32];   // [d][key row]  (transposed V)
    __shared__ _Float16 Pbuf[8][16 * 32]; // per-wave P tile (C->A relayout)
    const int bh = blockIdx.y;
    const int b  = bh >> 3;               // H = 8
    const int h  = bh & 7;
    const int t    = threadIdx.x;
    const int lane = t & 31;
    const int wave = t >> 5;
    const int i0 = blockIdx.x * 128 + wave * 16;
    const int lm = lane & 15;
    const int kg = lane >> 4;
    _Float16* P = Pbuf[wave];
    const float NEG = -__builtin_inff();

    // cooperative K/V staging coordinates (2 async b128 per thread per step)
    const int kjj  = t >> 3;              // 0..31  key row
    const int kseg = (t & 7) * 8;         // 0..56  d offset (halves)
    const int vdd  = t >> 2;              // 0..63  d row
    const int vseg = (t & 3) * 8;         // 0..24  key offset (halves)
    const _Float16* kgsrc = k_ws + ((size_t)bh * NN + kjj) * ND + kseg;
    const _Float16* vgsrc = vT_ws + ((size_t)bh * ND + vdd) * NN + vseg;

    // Q fragments for this wave's 16 rows (d = 0..31 and 32..63), kept live.
    const _Float16* qrow = q_ws + ((size_t)bh * NN + i0 + lm) * ND;
    const v16h qf0 = cat8(*(const v8h*)(qrow + kg * 16),
                          *(const v8h*)(qrow + kg * 16 + 8));
    const v16h qf1 = cat8(*(const v8h*)(qrow + 32 + kg * 16),
                          *(const v8h*)(qrow + 32 + kg * 16 + 8));

    float mrow[8], lrow[8], scl[8];
    v8f acc[4];
    v8f zero = {};
    #pragma unroll
    for (int r = 0; r < 8; ++r) { mrow[r] = NEG; lrow[r] = 0.0f; }
    #pragma unroll
    for (int dt = 0; dt < 4; ++dt) acc[dt] = zero;

    // prologue: stage j-block 0 into buffer 0
    async_copy_b128(&kS[0][kjj * 64 + kseg], kgsrc);
    async_copy_b128(&vS[0][vdd * 32 + vseg], vgsrc);
    wait_asynccnt0();
    __syncthreads();

    for (int j0 = 0; j0 < NN; j0 += 32) {
        const int cur = (j0 >> 5) & 1;
        // issue async staging of the NEXT key block while we compute
        if (j0 + 32 < NN) {
            async_copy_b128(&kS[cur ^ 1][kjj * 64 + kseg], kgsrc + (size_t)(j0 + 32) * ND);
            async_copy_b128(&vS[cur ^ 1][vdd * 32 + vseg], vgsrc + (j0 + 32));
        }

        // ---- scores: two 16x16 tiles from the staged K block ----
        v8f S0 = {}; v8f S1 = {};
        {
            const _Float16* krA = &kS[cur][lm * 64];          // jt = 0 rows
            const _Float16* krB = krA + 16 * 64;              // jt = 1 rows
            v16h kf;
            kf = cat8(*(const v8h*)(krA + kg * 16),      *(const v8h*)(krA + kg * 16 + 8));
            S0 = wmma_f16(qf0, kf, S0);
            kf = cat8(*(const v8h*)(krA + 32 + kg * 16), *(const v8h*)(krA + 32 + kg * 16 + 8));
            S0 = wmma_f16(qf1, kf, S0);
            kf = cat8(*(const v8h*)(krB + kg * 16),      *(const v8h*)(krB + kg * 16 + 8));
            S1 = wmma_f16(qf0, kf, S1);
            kf = cat8(*(const v8h*)(krB + 32 + kg * 16), *(const v8h*)(krB + 32 + kg * 16 + 8));
            S1 = wmma_f16(qf1, kf, S1);
        }

        // prefetch next block's bias/mul stream
        if (j0 + 32 < NN) {
            const size_t nb = ((size_t)(b * NN + i0 + lm) * NN + j0 + 32) * NH + h;
            __builtin_prefetch(attn_bias + nb, 0, 0);
            __builtin_prefetch(attn_mul + nb, 0, 0);
        }

        // ---- bias + mask + online softmax + post-softmax multiplier ----
        const int jc0 = j0 + lm;
        const int jc1 = jc0 + 16;
        #pragma unroll
        for (int r = 0; r < 8; ++r) {
            const int i = i0 + (kg ? r + 8 : r);
            const size_t base = (size_t)(b * NN + i) * NN;
            float a0 = S0[r] + attn_bias[(base + jc0) * NH + h];
            float a1 = S1[r] + attn_bias[(base + jc1) * NH + h];
            const int mk0 = attn_mask[base + jc0];
            const int mk1 = attn_mask[base + jc1];
            if (mk0) a0 = NEG;
            if (mk1) a1 = NEG;
            float bm = fmaxf(a0, a1);
            bm = fmaxf(bm, __shfl_xor(bm, 1, 32));
            bm = fmaxf(bm, __shfl_xor(bm, 2, 32));
            bm = fmaxf(bm, __shfl_xor(bm, 4, 32));
            bm = fmaxf(bm, __shfl_xor(bm, 8, 32));
            const float mn = fmaxf(mrow[r], bm);
            const float sc = (mrow[r] == NEG) ? 0.0f : __expf(mrow[r] - mn);
            const float e0 = mk0 ? 0.0f : __expf(a0 - mn);
            const float e1 = mk1 ? 0.0f : __expf(a1 - mn);
            float rs = e0 + e1;
            rs += __shfl_xor(rs, 1, 32);
            rs += __shfl_xor(rs, 2, 32);
            rs += __shfl_xor(rs, 4, 32);
            rs += __shfl_xor(rs, 8, 32);
            lrow[r] = lrow[r] * sc + rs;
            mrow[r] = mn;
            scl[r]  = sc;
            const float mu0 = attn_mul[(base + jc0) * NH + h];
            const float mu1 = attn_mul[(base + jc1) * NH + h];
            const int prow = kg ? r + 8 : r;
            P[prow * 32 + lm]      = (_Float16)(e0 * mu0);
            P[prow * 32 + 16 + lm] = (_Float16)(e1 * mu1);
        }

        // rescale running accumulator (per row)
        #pragma unroll
        for (int dt = 0; dt < 4; ++dt)
            #pragma unroll
            for (int r = 0; r < 8; ++r)
                acc[dt][r] *= scl[r];

        asm volatile("s_wait_dscnt 0" ::: "memory");

        // P (16x32) as A fragment, staged V block as four B fragments
        const _Float16* pp = P + lm * 32 + kg * 16;
        const v16h pf = cat8(*(const v8h*)pp, *(const v8h*)(pp + 8));
        #pragma unroll
        for (int dt = 0; dt < 4; ++dt) {
            const _Float16* vp = &vS[cur][(dt * 16 + lm) * 32 + kg * 16];
            const v16h vf = cat8(*(const v8h*)vp, *(const v8h*)(vp + 8));
            acc[dt] = wmma_f16(pf, vf, acc[dt]);
        }

        wait_asynccnt0();      // next K/V block landed in LDS
        __syncthreads();       // everyone done with cur + staging complete
    }

    // normalize and store attention output (f16, [B*N, F]) for final GEMM
    #pragma unroll
    for (int dt = 0; dt < 4; ++dt) {
        #pragma unroll
        for (int r = 0; r < 8; ++r) {
            const int i = i0 + (kg ? r + 8 : r);
            const float o = acc[dt][r] / lrow[r];
            ao_ws[(size_t)(b * NN + i) * NF + h * ND + dt * 16 + lm] = (_Float16)o;
        }
    }
}

// ---------------------------------------------------------------------------
// Kernel 3: output projection, f16 GEMM + f32 bias -> f32 out.
// ---------------------------------------------------------------------------
__global__ __launch_bounds__(256)
void out_proj_kernel(const _Float16* __restrict__ ao,
                     const float* __restrict__ Wo,
                     const float* __restrict__ bo,
                     float* __restrict__ out)
{
    __shared__ _Float16 As[64 * 32];
    __shared__ _Float16 Bs[64 * 32];
    const int R0 = blockIdx.x * 64;
    const int C0 = blockIdx.y * 64;
    const int t    = threadIdx.x;
    const int lane = t & 31;
    const int wave = t >> 5;
    const int msub = (wave >> 1) * 16;
    const int nsub = (wave & 1) * 32;
    const int lm = lane & 15;
    const int kg = lane >> 4;

    v8f c0 = {}; v8f c1 = {};
    const int arow = t >> 2;
    const int aks  = (t & 3) * 8;

    for (int kb = 0; kb < NF; kb += 32) {
        *(v8h*)(As + arow * 32 + aks) =
            *(const v8h*)(ao + (size_t)(R0 + arow) * NF + kb + aks);
        {
            const float* src = Wo + (size_t)(kb + aks) * NF + C0 + arow;
            _Float16* dst = Bs + arow * 32 + aks;
            #pragma unroll
            for (int j = 0; j < 8; ++j) dst[j] = (_Float16)src[(size_t)j * NF];
        }
        __syncthreads();
        const _Float16* ap = As + (msub + lm) * 32 + kg * 8;
        v16h a = cat8(*(const v8h*)ap, *(const v8h*)(ap + 16));
        const _Float16* bp0 = Bs + (nsub + lm) * 32 + kg * 8;
        v16h b0 = cat8(*(const v8h*)bp0, *(const v8h*)(bp0 + 16));
        const _Float16* bp1 = Bs + (nsub + 16 + lm) * 32 + kg * 8;
        v16h b1 = cat8(*(const v8h*)bp1, *(const v8h*)(bp1 + 16));
        c0 = wmma_f16(a, b0, c0);
        c1 = wmma_f16(a, b1, c1);
        __syncthreads();
    }

    #pragma unroll
    for (int nt = 0; nt < 2; ++nt) {
        const int colg = C0 + nsub + nt * 16 + lm;
        const float bb = bo[colg];
        v8f cc = nt ? c1 : c0;
        #pragma unroll
        for (int r = 0; r < 8; ++r) {
            const int rowg = R0 + msub + (kg ? r + 8 : r);
            out[(size_t)rowg * NF + colg] = cc[r] + bb;
        }
    }
}

// ---------------------------------------------------------------------------
extern "C" void kernel_launch(void* const* d_in, const int* in_sizes, int n_in,
                              void* d_out, int out_size, void* d_ws, size_t ws_size,
                              hipStream_t stream) {
    (void)in_sizes; (void)n_in; (void)out_size; (void)ws_size;
    const float* ndata     = (const float*)d_in[0];
    const float* attn_bias = (const float*)d_in[1];
    const int*   attn_mask = (const int*)  d_in[2];
    const float* attn_mul  = (const float*)d_in[3];
    const float* Wq = (const float*)d_in[4];
    const float* bq = (const float*)d_in[5];
    const float* Wk = (const float*)d_in[6];
    const float* bk = (const float*)d_in[7];
    const float* Wv = (const float*)d_in[8];
    const float* bv = (const float*)d_in[9];
    const float* Wo = (const float*)d_in[10];
    const float* bo = (const float*)d_in[11];
    float* out = (float*)d_out;

    _Float16* ws = (_Float16*)d_ws;
    const size_t QS = (size_t)NB * NH * NN * ND;   // 4M halves each
    _Float16* q_ws  = ws;
    _Float16* k_ws  = ws + QS;
    _Float16* vT_ws = ws + 2 * QS;
    _Float16* ao_ws = ws + 3 * QS;

    dim3 g1(NB * NN / 64, NF / 64, 3);
    qkv_proj_kernel<<<g1, 256, 0, stream>>>(ndata, Wq, bq, Wk, bk, Wv, bv,
                                            q_ws, k_ws, vT_ws);
    dim3 g2(NN / 128, NB * NH);
    attn_kernel<<<g2, 256, 0, stream>>>(attn_bias, attn_mask, attn_mul,
                                        q_ws, k_ws, vT_ws, ao_ws);
    dim3 g3(NB * NN / 64, NF / 64);
    out_proj_kernel<<<g3, 256, 0, stream>>>(ao_ws, Wo, bo, out);
}
